// AttentionHeteroConv_59820304498991
// MI455X (gfx1250) — compile-verified, matrix operand order checked by reading, not run on hardware
//
#include <hip/hip_runtime.h>
#include <hip/hip_bf16.h>

#define NU 50000
#define NI 50000
#define NE 800000
#define C  128
#define CH 10000   // node chunk for MHA phase

typedef __bf16 v16bf __attribute__((ext_vector_type(16)));
typedef __bf16 v8bf  __attribute__((ext_vector_type(8)));
typedef float  v8f   __attribute__((ext_vector_type(8)));

// ---------------- helpers ----------------
__device__ __forceinline__ unsigned f2ord(float f) {
    unsigned u = __float_as_uint(f);
    return (u & 0x80000000u) ? ~u : (u | 0x80000000u);
}
__device__ __forceinline__ float ord2f(unsigned u) {
    u = (u & 0x80000000u) ? (u & 0x7fffffffu) : ~u;
    return __uint_as_float(u);
}

// ---------------- weight f32 -> bf16 ----------------
__global__ void cvt_bf16_kernel(const float* __restrict__ in, __bf16* __restrict__ out, int n) {
    int i = blockIdx.x * 256 + threadIdx.x;
    if (i < n) out[i] = (__bf16)in[i];
}

// ---------------- generic GEMM: out[M,Nc] = A[M,128] @ Wb[Nc,128]^T + bias ----------------
// block = 256 threads (8 waves). Each wave: 16 rows x 64 cols (4 accumulators); one
// loaded+converted A fragment feeds 4 WMMAs per K-step. All B loads are a single per-lane
// base pointer + compile-time immediate offsets (tile: +4096B, k-step: +64B, half: +32B).
__global__ void gemm_wmma_kernel(const float* __restrict__ A,
                                 const __bf16* __restrict__ Wb,
                                 const float* __restrict__ bias,
                                 float* __restrict__ out,
                                 int M, int Nc) {
    const int wave = threadIdx.x >> 5;
    const int lane = threadIdx.x & 31;
    const int rowBase = blockIdx.x * 128 + wave * 16;
    const int colBase = blockIdx.y * 64;
    const int lrow = lane & 15;
    const int lhi  = lane >> 4;

    int arow = rowBase + lrow;
    if (arow >= M) arow = M - 1;                 // clamped load; store is guarded
    const float*  Abase = A  + (size_t)arow * 128 + lhi * 8;
    const __bf16* Bbase = Wb + (size_t)(colBase + lrow) * 128 + lhi * 8;

    v8f acc[4] = {};
#pragma unroll
    for (int kk = 0; kk < 128; kk += 32) {
        // ---- A fragment (convert f32 -> bf16 on the fly), reused by 4 WMMAs ----
        float4 a0 = *(const float4*)(Abase + kk);
        float4 a1 = *(const float4*)(Abase + kk + 4);
        float4 a2 = *(const float4*)(Abase + kk + 16);
        float4 a3 = *(const float4*)(Abase + kk + 20);
        v16bf af;
        af[0]  = (__bf16)a0.x; af[1]  = (__bf16)a0.y; af[2]  = (__bf16)a0.z; af[3]  = (__bf16)a0.w;
        af[4]  = (__bf16)a1.x; af[5]  = (__bf16)a1.y; af[6]  = (__bf16)a1.z; af[7]  = (__bf16)a1.w;
        af[8]  = (__bf16)a2.x; af[9]  = (__bf16)a2.y; af[10] = (__bf16)a2.z; af[11] = (__bf16)a2.w;
        af[12] = (__bf16)a3.x; af[13] = (__bf16)a3.y; af[14] = (__bf16)a3.z; af[15] = (__bf16)a3.w;

        // ---- 4 B fragments up front (base + immediate offsets -> one load clause) ----
        v16bf bf[4];
#pragma unroll
        for (int t = 0; t < 4; ++t) {
            v8bf b0 = *(const v8bf*)(Bbase + t * 2048 + kk);
            v8bf b1 = *(const v8bf*)(Bbase + t * 2048 + kk + 16);
#pragma unroll
            for (int i = 0; i < 8; ++i) { bf[t][i] = b0[i]; bf[t][i + 8] = b1[i]; }
        }
#pragma unroll
        for (int t = 0; t < 4; ++t)
            acc[t] = __builtin_amdgcn_wmma_f32_16x16x32_bf16(false, af, false, bf[t],
                                                             (short)0, acc[t], false, false);
    }

#pragma unroll
    for (int t = 0; t < 4; ++t) {
        const int col = colBase + t * 16 + lrow;
        const float bv = bias[col];
#pragma unroll
        for (int r = 0; r < 8; ++r) {
            int row = rowBase + r + 8 * lhi;     // C/D layout: lanes16-31 hold M=r+8
            if (row < M) out[(size_t)row * Nc + col] = acc[t][r] + bv;
        }
    }
}

// ---------------- aggregation ----------------
__global__ void init_aggr_kernel(float* __restrict__ aggr, float* __restrict__ cnt, int N) {
    int idx = blockIdx.x * 256 + threadIdx.x;
    if (idx >= N * C) return;
    int n = idx >> 7, c = idx & 127;
    float* base = aggr + (size_t)n * 512;
    ((unsigned*)base)[c]       = 0u;             // encoded -inf (max slot)
    ((unsigned*)base)[128 + c] = 0xFFFFFFFFu;    // encoded +inf (min slot)
    base[256 + c] = 0.f;                         // sum
    base[384 + c] = 0.f;                         // mean
    if (c == 0) cnt[n] = 0.f;
}

__global__ void count_kernel(const int* __restrict__ dst, float* __restrict__ cnt, int E) {
    int e = blockIdx.x * 256 + threadIdx.x;
    if (e < E) atomicAdd(&cnt[dst[e]], 1.0f);
}

__global__ void scatter_kernel(const float* __restrict__ h,
                               const int* __restrict__ src, const int* __restrict__ dst,
                               float* __restrict__ aggr, int E) {
    long long id = (long long)blockIdx.x * 256 + threadIdx.x;
    if (id >= (long long)E * C) return;
    int e = (int)(id >> 7), c = (int)(id & 127);
    int s = src[e], d = dst[e];
    float m = h[(size_t)s * C + c];
    float* base = aggr + (size_t)d * 512;
    atomicMax((unsigned*)base + c, f2ord(m));
    atomicMin((unsigned*)base + 128 + c, f2ord(m));
    atomicAdd(base + 256 + c, m);
}

__global__ void finalize_kernel(float* __restrict__ aggr, const float* __restrict__ cnt, int N) {
    int idx = blockIdx.x * 256 + threadIdx.x;
    if (idx >= N * C) return;
    int n = idx >> 7, c = idx & 127;
    float* base = aggr + (size_t)n * 512;
    float ct = cnt[n];
    bool has = ct > 0.f;
    float mx = has ? ord2f(((unsigned*)base)[c]) : 0.f;
    float mn = has ? ord2f(((unsigned*)base)[128 + c]) : 0.f;
    float s  = base[256 + c];
    base[c] = mx;
    base[128 + c] = mn;
    base[384 + c] = s / fmaxf(ct, 1.f);
}

// ---------------- build tokens [5, CHn, 128] ----------------
__global__ void tokens_kernel(const float* __restrict__ self, const float* __restrict__ aggr,
                              float* __restrict__ tok, int CHn) {
    int idx = blockIdx.x * 256 + threadIdx.x;
    if (idx >= CHn * C) return;
    int n = idx >> 7, c = idx & 127;
    const float* a = aggr + (size_t)n * 512;
    size_t stride = (size_t)CHn * C;
    tok[idx]              = self[idx];
    tok[stride + idx]     = a[c];
    tok[2 * stride + idx] = a[128 + c];
    tok[3 * stride + idx] = a[256 + c];
    tok[4 * stride + idx] = a[384 + c];
}

// ---------------- attention core: one wave per (node, head), lane = d ----------------
__global__ void attn_kernel(const float* __restrict__ qkv,  // [5, CHn, 384]
                            float* __restrict__ o,          // [5, CHn, 128]
                            int CHn) {
    int gid = blockIdx.x * (blockDim.x >> 5) + (threadIdx.x >> 5);
    int lane = threadIdx.x & 31;
    int n = gid >> 2, h = gid & 3;
    if (n >= CHn) return;
    const float scale = 0.17677669529663687f;   // 1/sqrt(32)
    size_t rs = (size_t)CHn * 384;
    const float* base = qkv + (size_t)n * 384 + h * 32 + lane;
    float q[5], k[5], v[5];
#pragma unroll
    for (int s = 0; s < 5; ++s) {
        q[s] = base[s * rs] * scale;
        k[s] = base[s * rs + 128];
        v[s] = base[s * rs + 256];
    }
    float sc[5][5];
#pragma unroll
    for (int s = 0; s < 5; ++s)
#pragma unroll
        for (int t = 0; t < 5; ++t) {
            float p = q[s] * k[t];
#pragma unroll
            for (int off = 16; off > 0; off >>= 1) p += __shfl_xor(p, off, 32);
            sc[s][t] = p;
        }
    size_t os = (size_t)CHn * C;
    float* ob = o + (size_t)n * C + h * 32 + lane;
#pragma unroll
    for (int s = 0; s < 5; ++s) {
        float m = sc[s][0];
#pragma unroll
        for (int t = 1; t < 5; ++t) m = fmaxf(m, sc[s][t]);
        float e[5], sum = 0.f;
#pragma unroll
        for (int t = 0; t < 5; ++t) { e[t] = __expf(sc[s][t] - m); sum += e[t]; }
        float inv = 1.f / sum, acc = 0.f;
#pragma unroll
        for (int t = 0; t < 5; ++t) acc += e[t] * inv * v[t];
        ob[s * os] = acc;
    }
}

// ---------------- out = self + mean_s(attn) ----------------
__global__ void combine_kernel(const float* __restrict__ self, const float* __restrict__ attn,
                               float* __restrict__ outp, int CHn) {
    int idx = blockIdx.x * 256 + threadIdx.x;
    if (idx >= CHn * C) return;
    size_t stride = (size_t)CHn * C;
    float a = 0.f;
#pragma unroll
    for (int s = 0; s < 5; ++s) a += attn[s * stride + idx];
    outp[idx] = self[idx] + a * 0.2f;
}

// ---------------- orchestration ----------------
extern "C" void kernel_launch(void* const* d_in, const int* in_sizes, int n_in,
                              void* d_out, int out_size, void* d_ws, size_t ws_size,
                              hipStream_t stream) {
    (void)in_sizes; (void)n_in; (void)out_size; (void)ws_size;
    const float* x_user     = (const float*)d_in[0];
    const float* x_item     = (const float*)d_in[1];
    const int*   ei_u2i     = (const int*)d_in[2];   // [2,E]: row0=user(src), row1=item(dst)
    const int*   ei_i2u     = (const int*)d_in[3];   // [2,E]: row0=item(src), row1=user(dst)
    const float* W_nb       = (const float*)d_in[4];
    const float* b_nb       = (const float*)d_in[5];
    const float* W_self     = (const float*)d_in[6];
    const float* b_self     = (const float*)d_in[7];
    const float* in_proj_w  = (const float*)d_in[8];
    const float* in_proj_b  = (const float*)d_in[9];
    const float* out_proj_w = (const float*)d_in[10];
    const float* out_proj_b = (const float*)d_in[11];
    float* outp = (float*)d_out;

    char* ws = (char*)d_ws;
    size_t off = 0;
    auto take = [&](size_t bytes) { size_t o = off; off = (off + bytes + 255) & ~(size_t)255; return o; };
    float*  h_user    = (float*)(ws + take((size_t)NU * C * 4));   // reused as self_user
    float*  h_item    = (float*)(ws + take((size_t)NI * C * 4));   // reused as self_item
    float*  aggr_user = (float*)(ws + take((size_t)NU * 4 * C * 4));
    float*  aggr_item = (float*)(ws + take((size_t)NI * 4 * C * 4));
    float*  cnt_user  = (float*)(ws + take((size_t)NU * 4));
    float*  cnt_item  = (float*)(ws + take((size_t)NI * 4));
    __bf16* wnb_bf    = (__bf16*)(ws + take((size_t)C * C * 2));
    __bf16* wself_bf  = (__bf16*)(ws + take((size_t)C * C * 2));
    __bf16* winp_bf   = (__bf16*)(ws + take((size_t)3 * C * C * 2));
    __bf16* woutp_bf  = (__bf16*)(ws + take((size_t)C * C * 2));
    float*  tok_buf   = (float*)(ws + take((size_t)5 * CH * C * 4));      // tokens, then attn-o
    float*  qkv_buf   = (float*)(ws + take((size_t)5 * CH * 3 * C * 4));  // qkv, then out_proj result

    auto cvt = [&](const float* src, __bf16* dst, int n) {
        cvt_bf16_kernel<<<(n + 255) / 256, 256, 0, stream>>>(src, dst, n);
    };
    auto gemm = [&](const float* A, const __bf16* Wb, const float* bias, float* O, int M, int Nc) {
        dim3 g((M + 127) / 128, Nc / 64);
        gemm_wmma_kernel<<<g, 256, 0, stream>>>(A, Wb, bias, O, M, Nc);
    };

    // 1) weights -> bf16
    cvt(W_nb, wnb_bf, C * C);
    cvt(W_self, wself_bf, C * C);
    cvt(in_proj_w, winp_bf, 3 * C * C);
    cvt(out_proj_w, woutp_bf, C * C);

    // 2) neighbor linear (WMMA GEMMs)
    gemm(x_user, wnb_bf, b_nb, h_user, NU, C);
    gemm(x_item, wnb_bf, b_nb, h_item, NI, C);

    // 3) segmented max/min/sum/mean via atomics
    int nc_blocks = (NU * C + 255) / 256;
    init_aggr_kernel<<<nc_blocks, 256, 0, stream>>>(aggr_item, cnt_item, NI);
    init_aggr_kernel<<<nc_blocks, 256, 0, stream>>>(aggr_user, cnt_user, NU);
    int ec_blocks = (int)(((long long)NE * C + 255) / 256);
    // user->item aggregation uses h_user and ei_u2i
    scatter_kernel<<<ec_blocks, 256, 0, stream>>>(h_user, ei_u2i, ei_u2i + NE, aggr_item, NE);
    count_kernel<<<(NE + 255) / 256, 256, 0, stream>>>(ei_u2i + NE, cnt_item, NE);
    // item->user aggregation uses h_item and ei_i2u
    scatter_kernel<<<ec_blocks, 256, 0, stream>>>(h_item, ei_i2u, ei_i2u + NE, aggr_user, NE);
    count_kernel<<<(NE + 255) / 256, 256, 0, stream>>>(ei_i2u + NE, cnt_user, NE);
    finalize_kernel<<<nc_blocks, 256, 0, stream>>>(aggr_item, cnt_item, NI);
    finalize_kernel<<<nc_blocks, 256, 0, stream>>>(aggr_user, cnt_user, NU);

    // 4) self linear (h buffers no longer needed -> reuse as self features)
    gemm(x_user, wself_bf, b_self, h_user, NU, C);
    gemm(x_item, wself_bf, b_self, h_item, NI, C);

    // 5) MHA per type, chunked over nodes
    for (int type = 0; type < 2; ++type) {
        const float* self = (type == 0) ? h_user : h_item;
        const float* aggr = (type == 0) ? aggr_user : aggr_item;
        const int    Nn   = (type == 0) ? NU : NI;
        float* obase = outp + (size_t)type * NU * C;
        for (int nb = 0; nb < Nn; nb += CH) {
            const float* selfc = self + (size_t)nb * C;
            const float* aggrc = aggr + (size_t)nb * 4 * C;
            int chn = CH;  // NU,NI divisible by CH
            int cc_blocks = (chn * C + 255) / 256;
            tokens_kernel<<<cc_blocks, 256, 0, stream>>>(selfc, aggrc, tok_buf, chn);
            gemm(tok_buf, winp_bf, in_proj_b, qkv_buf, 5 * chn, 3 * C);        // qkv
            attn_kernel<<<(chn * 4 + 7) / 8, 256, 0, stream>>>(qkv_buf, tok_buf, chn); // o -> tok_buf
            gemm(tok_buf, woutp_bf, out_proj_b, qkv_buf, 5 * chn, C);          // out_proj -> qkv_buf
            combine_kernel<<<cc_blocks, 256, 0, stream>>>(selfc, qkv_buf, obase + (size_t)nb * C, chn);
        }
    }
}
